// BvGENConvBig_85633057948335
// MI455X (gfx1250) — 8-bit matrix operand order repaired and checked by reading, compiled, not run on hardware
//
#include <hip/hip_runtime.h>
#include <hip/hip_bf16.h>

#define NN 50000
#define EE 625000
#define CC 128
#define LL 4
#define HH 256

typedef __attribute__((ext_vector_type(8))) int v8i;
typedef __attribute__((ext_vector_type(4))) int v4i;
typedef __attribute__((ext_vector_type(4))) unsigned int v4u;

#if defined(__has_builtin)
#if __has_builtin(__builtin_amdgcn_tensor_load_to_lds) && __has_builtin(__builtin_amdgcn_s_wait_tensorcnt)
#define HAVE_TDM 1
#endif
#if __has_builtin(__builtin_amdgcn_sdot4)
#define HAVE_DOT4 1
#endif
#endif

__device__ __forceinline__ int fq_i(float x) {
    float r = rintf(x * 16.0f);
    r = fminf(fmaxf(r, -128.0f), 127.0f);
    return (int)r;
}
__device__ __forceinline__ int rhe16(int acc) {
    int q = acc >> 4;
    int r = acc & 15;
    q += (r > 8) ? 1 : ((r == 8) ? (q & 1) : 0);
    return q;
}
__device__ __forceinline__ int clampq(int v) {
    v = v < -128 ? -128 : v;
    return v > 127 ? 127 : v;
}
__device__ __forceinline__ int pack4(int a, int b, int c, int d) {
    return (a & 0xff) | ((b & 0xff) << 8) | ((c & 0xff) << 16) | ((d & 0xff) << 24);
}
__device__ __forceinline__ float qstepf(float accf, float biasf, float lo) {
    float z = rintf(accf * 0.0625f + biasf);
    return fminf(fmaxf(z, lo), 127.0f);
}

#ifdef HAVE_TDM
__device__ __forceinline__ void tdm_load_1d(unsigned lds_off, const void* gptr, unsigned bytes) {
    unsigned n = bytes >> 3;
    unsigned long long ga = (unsigned long long)(uintptr_t)gptr;
    v4u g0;
    g0.x = 1u;
    g0.y = lds_off;
    g0.z = (unsigned)(ga & 0xffffffffu);
    g0.w = (unsigned)((ga >> 32) & 0x1ffffffu) | (2u << 30);
    v8i g1;
    g1[0] = 3 << 16;
    g1[1] = (int)((n & 0xffffu) << 16);
    g1[2] = (int)(((n >> 16) & 0xffffu) | (1u << 16));
    g1[3] = (int)((n & 0xffffu) << 16);
    g1[4] = 0;
    g1[5] = (int)n;
    g1[6] = 0;
    g1[7] = 0;
    v4i z4 = {0, 0, 0, 0};
    v8i z8 = {0, 0, 0, 0, 0, 0, 0, 0};
    __builtin_amdgcn_tensor_load_to_lds(g0, g1, z4, z4, z8, 0);
}
#endif

__global__ void k_wq(const float* __restrict__ W1, const float* __restrict__ W2,
                     signed char* __restrict__ W1T, signed char* __restrict__ W2T) {
    int idx = blockIdx.x * blockDim.x + threadIdx.x;
    const int total = LL * CC * HH;
    if (idx >= total) return;
    {
        int l = idx / (CC * HH);
        int r = idx - l * (CC * HH);
        int c = r / HH;
        int h = r - c * HH;
        W1T[(size_t)l * CC * HH + (size_t)h * CC + c] = (signed char)fq_i(W1[idx]);
    }
    {
        int l = idx / (HH * CC);
        int r = idx - l * (HH * CC);
        int h = r / CC;
        int c = r - h * CC;
        W2T[(size_t)l * HH * CC + (size_t)c * HH + h] = (signed char)fq_i(W2[idx]);
    }
}

__global__ void k_eaq(const float* __restrict__ ea, int* __restrict__ eaq) {
    int e = blockIdx.x * blockDim.x + threadIdx.x;
    if (e >= EE) return;
    float4 a4 = ((const float4*)ea)[e];
    eaq[e] = pack4(fq_i(a4.x), fq_i(a4.y), fq_i(a4.z), fq_i(a4.w));
}

__global__ void k_prep(const float* __restrict__ eW, const float* __restrict__ eb,
                       const float* __restrict__ bnw, const float* __restrict__ bnb,
                       int* __restrict__ p_wp, float* __restrict__ p_bf,
                       float* __restrict__ p_g16, float* __restrict__ p_be) {
    int c = threadIdx.x;
    if (c >= CC) return;
    p_wp[c] = pack4(fq_i(eW[c]), fq_i(eW[CC + c]), fq_i(eW[2 * CC + c]), fq_i(eW[3 * CC + c]));
    p_bf[c] = (float)fq_i(eb[c]);
    p_g16[c] = (float)fq_i(bnw[c]) * 0.0625f;
    p_be[c] = (float)fq_i(bnb[c]);
}

__global__ void k_node_enc(const float* __restrict__ x, const float* __restrict__ nW,
                           const float* __restrict__ nb, const float* __restrict__ bnw,
                           const float* __restrict__ bnb, short* __restrict__ h16) {
    int idx = blockIdx.x * blockDim.x + threadIdx.x;
    if (idx >= NN * CC) return;
    int n = idx >> 7, c = idx & 127;
    int acc = 0;
#pragma unroll
    for (int k = 0; k < 3; ++k)
        acc += fq_i(x[n * 3 + k]) * fq_i(nW[k * CC + c]);
    int z = clampq(rhe16(acc) + fq_i(nb[c]));
    int h = clampq(rhe16(z * fq_i(bnw[c])) + fq_i(bnb[c]));
    h16[idx] = (short)h;
}

__global__ __launch_bounds__(256) void k_msg(const int* __restrict__ ei,
                                             const int* __restrict__ eaq,
                                             const int* __restrict__ p_wp,
                                             const float* __restrict__ p_bf,
                                             const float* __restrict__ p_g16,
                                             const float* __restrict__ p_be,
                                             const short* __restrict__ h16,
                                             int* __restrict__ agg) {
    __shared__ int s_wp[CC];
    __shared__ float s_bf[CC], s_g16[CC], s_be[CC];
    if (threadIdx.x < CC) {
        int t = threadIdx.x;
        s_wp[t] = p_wp[t];
        s_bf[t] = p_bf[t];
        s_g16[t] = p_g16[t];
        s_be[t] = p_be[t];
    }
    __syncthreads();

    int wave = threadIdx.x >> 5, lane = threadIdx.x & 31;
    int e = blockIdx.x * 8 + wave;
    if (e >= EE) return;
    int src = ei[e];
    int tgt = ei[EE + e];
    int aq_pk = eaq[e];

    int cbase = lane * 4;
    const short* hr = h16 + (size_t)src * CC + cbase;
    int* ar = agg + (size_t)tgt * CC + cbase;
    short4 hv4 = *(const short4*)hr;
    int hv[4] = {hv4.x, hv4.y, hv4.z, hv4.w};
#pragma unroll
    for (int j = 0; j < 4; ++j) {
        int c = cbase + j;
#ifdef HAVE_DOT4
        int acc = __builtin_amdgcn_sdot4(aq_pk, s_wp[c], 0, false);
#else
        int wpk = s_wp[c];
        int acc = ((int)(signed char)(aq_pk)) * ((int)(signed char)(wpk)) +
                  ((int)(signed char)(aq_pk >> 8)) * ((int)(signed char)(wpk >> 8)) +
                  ((int)(signed char)(aq_pk >> 16)) * ((int)(signed char)(wpk >> 16)) +
                  ((int)(signed char)(aq_pk >> 24)) * ((int)(signed char)(wpk >> 24));
#endif
        float z = qstepf((float)acc, s_bf[c], -128.0f);
        float t = rintf(z * s_g16[c] + s_be[c]);
        t = fminf(fmaxf(t, -128.0f), 127.0f);
        int eint = (int)t;
        int m = clampq(hv[j]) + eint;
        m = m < 0 ? 0 : (m > 127 ? 127 : m);
        atomicAdd(ar + j, m);
    }
}

__global__ __launch_bounds__(256) void k_mlp(const int* __restrict__ agg,
                                             const signed char* __restrict__ w1l,
                                             const signed char* __restrict__ w2l,
                                             const float* __restrict__ b1l,
                                             const float* __restrict__ gl,
                                             const float* __restrict__ bel,
                                             const float* __restrict__ b2l,
                                             short* __restrict__ h16) {
    extern __shared__ signed char smem[];
    signed char* a2 = smem + (threadIdx.x >> 5) * (16 * HH);
    signed char* w1s = smem + 8 * 16 * HH;
    signed char* w2s = w1s + CC * HH;

#ifdef HAVE_TDM
    if (threadIdx.x == 0) {
        tdm_load_1d((unsigned)(uintptr_t)w1s, w1l, CC * HH);
        tdm_load_1d((unsigned)(uintptr_t)w2s, w2l, HH * CC);
        __builtin_amdgcn_s_wait_tensorcnt(0);
    }
#else
    for (int t = threadIdx.x; t < (CC * HH) / 16; t += 256) {
        ((int4*)w1s)[t] = ((const int4*)w1l)[t];
        ((int4*)w2s)[t] = ((const int4*)w2l)[t];
    }
#endif
    __syncthreads();

    int wave = threadIdx.x >> 5, lane = threadIdx.x & 31;
    int half = lane >> 4, mlane = lane & 15;
    int rowbase = blockIdx.x * 128 + wave * 16;

    int arow = rowbase + mlane;
    if (arow >= NN) arow = NN - 1;
    const int* ap = agg + (size_t)arow * CC;
    v8i a1[2];
#pragma unroll
    for (int kt = 0; kt < 2; ++kt) {
#pragma unroll
        for (int j = 0; j < 8; ++j) {
            int kq = kt * 64 + (j & 3) * 4 + (j >> 2) * 32 + half * 16;
            int4 t = *(const int4*)(ap + kq);
            int x0 = t.x > 127 ? 127 : (t.x < 0 ? 0 : t.x);
            int x1 = t.y > 127 ? 127 : (t.y < 0 ? 0 : t.y);
            int x2 = t.z > 127 ? 127 : (t.z < 0 ? 0 : t.z);
            int x3 = t.w > 127 ? 127 : (t.w < 0 ? 0 : t.w);
            a1[kt][j] = pack4(x0, x1, x2, x3);
        }
    }

#pragma unroll 1
    for (int nt = 0; nt < 16; ++nt) {
        int col = nt * 16 + mlane;
        const signed char* wp = w1s + (size_t)col * CC;
        v8i bf[2];
#pragma unroll
        for (int kt = 0; kt < 2; ++kt) {
            int kb = kt * 64 + half * 16;
            int4 lo = *(const int4*)(wp + kb);
            int4 hi = *(const int4*)(wp + kb + 32);
            bf[kt][0] = lo.x; bf[kt][1] = lo.y; bf[kt][2] = lo.z; bf[kt][3] = lo.w;
            bf[kt][4] = hi.x; bf[kt][5] = hi.y; bf[kt][6] = hi.z; bf[kt][7] = hi.w;
        }
        v8i acc = {0, 0, 0, 0, 0, 0, 0, 0};
        acc = __builtin_amdgcn_wmma_i32_16x16x64_iu8(true, a1[0], true, bf[0], acc, false, false);
        acc = __builtin_amdgcn_wmma_i32_16x16x64_iu8(true, a1[1], true, bf[1], acc, false, false);

        float b1f = (float)fq_i(b1l[col]);
        float g16 = (float)fq_i(gl[col]) * 0.0625f;
        float bef = (float)fq_i(bel[col]);
#pragma unroll
        for (int r = 0; r < 8; ++r) {
            int m = r + half * 8;
            float z = qstepf((float)acc[r], b1f, -128.0f);
            float t = rintf(z * g16 + bef);
            t = fminf(fmaxf(t, 0.0f), 127.0f);
            a2[m * HH + nt * 16 + mlane] = (signed char)(int)t;
        }
    }

#pragma unroll 1
    for (int nt = 0; nt < 8; ++nt) {
        int col = nt * 16 + mlane;
        const signed char* wp = w2s + (size_t)col * HH;
        v8i acc = {0, 0, 0, 0, 0, 0, 0, 0};
#pragma unroll
        for (int kt = 0; kt < 4; ++kt) {
            v8i a;
            int abase = kt * 64 + half * 16;
#pragma unroll
            for (int p = 0; p < 4; ++p) {
                uint2 t = *(const uint2*)(a2 + mlane * HH + abase + (p & 1) * 8 + (p >> 1) * 32);
                a[2 * p] = (int)t.x;
                a[2 * p + 1] = (int)t.y;
            }
            int kb = kt * 64 + half * 16;
            int4 lo = *(const int4*)(wp + kb);
            int4 hi = *(const int4*)(wp + kb + 32);
            v8i bfv;
            bfv[0] = lo.x; bfv[1] = lo.y; bfv[2] = lo.z; bfv[3] = lo.w;
            bfv[4] = hi.x; bfv[5] = hi.y; bfv[6] = hi.z; bfv[7] = hi.w;
            acc = __builtin_amdgcn_wmma_i32_16x16x64_iu8(true, a, true, bfv, acc, false, false);
        }
        float b2f = (float)fq_i(b2l[col]);
#pragma unroll
        for (int r = 0; r < 8; ++r) {
            int m = r + half * 8;
            int row = rowbase + m;
            if (row < NN) {
                float z = qstepf((float)acc[r], b2f, -128.0f);
                size_t off = (size_t)row * CC + col;
                h16[off] = (short)((int)h16[off] + (int)z);
            }
        }
    }
}

__global__ void k_pool(const short* __restrict__ h16, int* __restrict__ colsum) {
    int c = threadIdx.x & 127;
    int n0 = blockIdx.x * 2 + (threadIdx.x >> 7);
    int local = 0;
    for (int n = n0; n < NN; n += gridDim.x * 2)
        local += (int)h16[(size_t)n * CC + c];
    atomicAdd(&colsum[c], local);
}

__global__ void k_final(const int* __restrict__ colsum, const float* __restrict__ fcW,
                        const float* __restrict__ fcb, float* __restrict__ out) {
    __shared__ int red[128];
    int c = threadIdx.x;
    float pooled = (float)colsum[c] / (16.0f * (float)NN);
    int pq = fq_i(pooled);
    red[c] = pq * fq_i(fcW[c]);
    __syncthreads();
    for (int s = 64; s > 0; s >>= 1) {
        if (c < s) red[c] += red[c + s];
        __syncthreads();
    }
    if (c == 0) {
        int oq = clampq(rhe16(red[0]) + fq_i(fcb[0]));
        float ov = (float)oq / 16.0f;
        float sg = 1.0f / (1.0f + expf(-ov));
        out[0] = (float)fq_i(sg) / 16.0f;
    }
}

extern "C" void kernel_launch(void* const* d_in, const int* in_sizes, int n_in,
                              void* d_out, int out_size, void* d_ws, size_t ws_size,
                              hipStream_t stream) {
    const float* x    = (const float*)d_in[0];
    const float* ea   = (const float*)d_in[1];
    const int*   ei   = (const int*)d_in[2];
    const float* nW   = (const float*)d_in[3];
    const float* nb   = (const float*)d_in[4];
    const float* eW   = (const float*)d_in[5];
    const float* eb   = (const float*)d_in[6];
    const float* nbnw = (const float*)d_in[7];
    const float* nbnb = (const float*)d_in[8];
    const float* ebnw = (const float*)d_in[9];
    const float* ebnb = (const float*)d_in[10];
    const float* W1   = (const float*)d_in[11];
    const float* b1   = (const float*)d_in[12];
    const float* bnw  = (const float*)d_in[13];
    const float* bnb  = (const float*)d_in[14];
    const float* W2   = (const float*)d_in[15];
    const float* b2   = (const float*)d_in[16];
    const float* fcW  = (const float*)d_in[17];
    const float* fcb  = (const float*)d_in[18];
    float* out = (float*)d_out;

    char* base = (char*)d_ws;
    size_t o = 0;
    auto carve = [&](size_t bytes) {
        void* p = base + o;
        o = (o + bytes + 255) & ~(size_t)255;
        return p;
    };
    short*       h16    = (short*)carve((size_t)NN * CC * 2);
    int*         agg    = (int*)carve((size_t)NN * CC * 4);
    signed char* W1T    = (signed char*)carve((size_t)LL * CC * HH);
    signed char* W2T    = (signed char*)carve((size_t)LL * HH * CC);
    int*         eaq    = (int*)carve((size_t)EE * 4);
    int*         p_wp   = (int*)carve((size_t)CC * 4);
    float*       p_bf   = (float*)carve((size_t)CC * 4);
    float*       p_g16  = (float*)carve((size_t)CC * 4);
    float*       p_be   = (float*)carve((size_t)CC * 4);
    int*         colsum = (int*)carve((size_t)CC * 4);
    (void)ws_size; (void)in_sizes; (void)n_in; (void)out_size;

    k_wq<<<(LL * CC * HH + 255) / 256, 256, 0, stream>>>(W1, W2, W1T, W2T);
    k_eaq<<<(EE + 255) / 256, 256, 0, stream>>>(ea, eaq);
    k_prep<<<1, 128, 0, stream>>>(eW, eb, ebnw, ebnb, p_wp, p_bf, p_g16, p_be);
    k_node_enc<<<(NN * CC + 255) / 256, 256, 0, stream>>>(x, nW, nb, nbnw, nbnb, h16);

    const size_t mlp_lds = 8 * 16 * HH + CC * HH + HH * CC;
    for (int l = 0; l < LL; ++l) {
        (void)hipMemsetAsync(agg, 0, (size_t)NN * CC * 4, stream);
        k_msg<<<(EE + 7) / 8, 256, 0, stream>>>(ei, eaq, p_wp, p_bf, p_g16, p_be, h16, agg);
        k_mlp<<<(NN + 127) / 128, 256, mlp_lds, stream>>>(
            agg,
            W1T + (size_t)l * CC * HH, W2T + (size_t)l * HH * CC,
            b1 + (size_t)l * HH, bnw + (size_t)l * HH, bnb + (size_t)l * HH,
            b2 + (size_t)l * CC, h16);
    }

    (void)hipMemsetAsync(colsum, 0, CC * 4, stream);
    k_pool<<<256, 256, 0, stream>>>(h16, colsum);
    k_final<<<1, 128, 0, stream>>>(colsum, fcW, fcb, out);
}
